// GAE_encoder_36429912605472
// MI455X (gfx1250) — matrix operand order repaired and checked by reading, not compiled, not run on hardware
//
#include <hip/hip_runtime.h>

typedef __attribute__((ext_vector_type(16))) _Float16 v16h;
typedef __attribute__((ext_vector_type(8)))  _Float16 v8h;
typedef __attribute__((ext_vector_type(4)))  _Float16 v4h;
typedef __attribute__((ext_vector_type(8)))  float    v8f;
typedef __attribute__((ext_vector_type(4)))  float    v4f;

#define BN_EPS 1e-5f

// ---------------------------------------------------------------- utilities

__global__ void zero_f32(float* __restrict__ p, long n) {
  long i = (long)blockIdx.x * blockDim.x + threadIdx.x;
  long stride = (long)gridDim.x * blockDim.x;
  for (; i < n; i += stride) p[i] = 0.f;
}

__global__ void zero_f32_v4(float* __restrict__ p, long nvec) {
  long i = (long)blockIdx.x * blockDim.x + threadIdx.x;
  long stride = (long)gridDim.x * blockDim.x;
  v4f z = {};
  for (; i < nvec; i += stride) *(v4f*)(p + i * 4) = z;
}

// per-column sum / sum-of-squares over an [M x C] row-major matrix.
// blockDim.x == C (64 or 128); one row per block iteration; partials
// merged with a single pair of float atomics per thread.
__global__ void col_stats(const float* __restrict__ x, int M, int C,
                          float* __restrict__ sum, float* __restrict__ sumsq) {
  int c = threadIdx.x;
  float s = 0.f, q = 0.f;
  for (int r = blockIdx.x; r < M; r += gridDim.x) {
    float v = x[(long)r * C + c];
    s += v;
    q += v * v;
  }
  atomicAdd(&sum[c], s);
  atomicAdd(&sumsq[c], q);
}

// fold BN + affine into one FMA: y = scale[c]*x + shift[c]
__global__ void bn_finalize(const float* __restrict__ sum, const float* __restrict__ sumsq,
                            const float* __restrict__ gamma, const float* __restrict__ beta,
                            float* __restrict__ scale, float* __restrict__ shift,
                            int C, float invM) {
  int c = threadIdx.x;
  if (c < C) {
    float mu  = sum[c] * invM;
    float var = sumsq[c] * invM - mu * mu;
    float s   = gamma[c] * rsqrtf(var + BN_EPS);
    scale[c] = s;
    shift[c] = beta[c] - mu * s;
  }
}

// xn = scale[c]*x + shift[c], C = 128, float4-vectorized (channel base 4-aligned)
__global__ void apply_affine128(const float* __restrict__ x,
                                const float* __restrict__ scale, const float* __restrict__ shift,
                                float* __restrict__ o, long nvec) {
  long i = (long)blockIdx.x * blockDim.x + threadIdx.x;
  long stride = (long)gridDim.x * blockDim.x;
  for (; i < nvec; i += stride) {
    long base = i * 4;
    int c = (int)(base & 127);
    v4f xv = *(const v4f*)(x + base);
    v4f sc = *(const v4f*)(scale + c);
    v4f sh = *(const v4f*)(shift + c);
    *(v4f*)(o + base) = xv * sc + sh;
  }
}

// out = tanh(scale[c]*h + shift[c]), C = 64, float4-vectorized
__global__ void apply_affine_tanh64(const float* __restrict__ h,
                                    const float* __restrict__ scale, const float* __restrict__ shift,
                                    float* __restrict__ out, long nvec) {
  long i = (long)blockIdx.x * blockDim.x + threadIdx.x;
  long stride = (long)gridDim.x * blockDim.x;
  for (; i < nvec; i += stride) {
    long base = i * 4;
    int c = (int)(base & 63);
    v4f hv = *(const v4f*)(h + base);
    v4f sc = *(const v4f*)(scale + c);
    v4f sh = *(const v4f*)(shift + c);
    v4f v  = hv * sc + sh;
    v4f r;
#pragma unroll
    for (int j = 0; j < 4; ++j) r[j] = tanhf(v[j]);
    *(v4f*)(out + base) = r;
  }
}

// GIN sum-aggregation: one 128-thread block per edge; lane c gathers channel c
// of the source row and atomically adds into the destination row. Both the
// gather source (xn, 51 MB) and the scatter target (agg, 51 MB) are L2-resident
// on the 192 MB L2, so this is an L2-atomic-throughput problem, not HBM.
__global__ void scatter_add128(const float* __restrict__ xn, const int* __restrict__ src,
                               const int* __restrict__ dst, float* __restrict__ agg, int E) {
  int e = blockIdx.x;
  if (e < E) {
    int c = threadIdx.x;
    int s = src[e], d = dst[e];
    atomicAdd(&agg[(long)d * 128 + c], xn[(long)s * 128 + c]);
  }
}

// h16 = f16(xn + agg), float4 in -> 4xf16 (b64) out
__global__ void add_to_f16(const float* __restrict__ a, const float* __restrict__ b,
                           _Float16* __restrict__ o, long nvec) {
  long i = (long)blockIdx.x * blockDim.x + threadIdx.x;
  long stride = (long)gridDim.x * blockDim.x;
  for (; i < nvec; i += stride) {
    long base = i * 4;
    v4f s = *(const v4f*)(a + base) + *(const v4f*)(b + base);
    v4h hv;
#pragma unroll
    for (int j = 0; j < 4; ++j) hv[j] = (_Float16)s[j];
    *(v4h*)(o + base) = hv;
  }
}

// ---------------------------------------------------- weight pre-swizzle
// Convert W [KD x NC] f32 row-major into per-lane WMMA B-fragment order so
// that a wave's B fragment (16 halfs/lane) is one contiguous 32-byte load:
//   Bs[(((nt*NKS)+ks)*32 + lane)*16 + j]
// with the v16h element mapping of v_wmma_f32_16x16x32_f16's B operand
// (CDNA5 ISA §7.12.2, 16-bit B 32x16):
//   lane<16 -> N = nt*16+lane,   element j -> K = ks*32 + (j&7) + (j>>3)*16
//   lane>=16 -> N = nt*16+lane-16, K += 8
__global__ void swizzle_B(const float* __restrict__ W, _Float16* __restrict__ Bs,
                          int KD, int NC) {
  const int NKS = KD >> 5;
  long total = (long)(NC >> 4) * NKS * 32 * 16;
  long i = (long)blockIdx.x * blockDim.x + threadIdx.x;
  long stride = (long)gridDim.x * blockDim.x;
  for (; i < total; i += stride) {
    int j    = (int)(i & 15);
    int lane = (int)((i >> 4) & 31);
    long t   = i >> 9;                  // nt*NKS + ks
    int ks   = (int)(t % NKS);
    int nt   = (int)(t / NKS);
    int half16 = lane >> 4;
    int ncol = nt * 16 + (lane & 15);
    int K    = ks * 32 + (j & 7) + half16 * 8 + (j >> 3) * 16;
    Bs[i] = (_Float16)W[(long)K * NC + ncol];
  }
}

// ------------------------------------------------------------- WMMA GEMM
// D = relu?(A @ B + bias). A: [M x KD] f16 row-major, Bs: pre-swizzled
// fragments, bias per column. Each wave owns ONE 16-wide column tile and
// iterates over row tiles; B fragments live in registers for the whole
// kernel. KD/NC are compile-time so all loop loads/stores use immediate
// instruction offsets off two per-lane pointers that advance by a constant
// stride -- no address arithmetic in the steady-state loop.
//
// A-fragment (16-bit A 16x32): per lane two contiguous 8-half runs:
//   af[0..7]  = A[row, kt + half16*8      .. +7]
//   af[8..15] = A[row, kt + 16 + half16*8 .. +7]
// C/D f32 16x16: VGPR r -> (M=m0+r+half16*8, N=ncol)
template <int KD, int NC, bool RELU, typename OUT_T>
__global__ void wmma_gemm(const _Float16* __restrict__ A, const _Float16* __restrict__ Bs,
                          const float* __restrict__ bias, OUT_T* __restrict__ out, int M) {
  constexpr int NKS   = KD / 32;
  constexpr int NTILE = NC / 16;
  const int lane   = threadIdx.x & 31;
  const int wave   = threadIdx.x >> 5;
  const int wpb    = blockDim.x >> 5;
  const int gw     = blockIdx.x * wpb + wave;
  const int totalW = gridDim.x * wpb;        // launched as a multiple of NTILE
  const int nt     = gw % NTILE;
  const int mt0    = gw / NTILE;
  const int mstep  = totalW / NTILE;
  const int half16 = lane >> 4;
  const int l15    = lane & 15;
  const int ncol   = nt * 16 + l15;

  // B fragments for all k-steps: one 32B contiguous load per k-step.
  v16h bf[NKS];
#pragma unroll
  for (int ks = 0; ks < NKS; ++ks)
    bf[ks] = *(const v16h*)(Bs + ((nt * NKS + ks) * 32 + lane) * 16);

  const float bv = bias[ncol];
  const int mtiles = M >> 4;

  const _Float16* ap = A + (long)(mt0 * 16 + l15) * KD + half16 * 8;
  OUT_T*           op = out + (long)(mt0 * 16 + half16 * 8) * NC + ncol;
  const long aInc = (long)mstep * 16 * KD;
  const long oInc = (long)mstep * 16 * NC;

  for (int mt = mt0; mt < mtiles; mt += mstep) {
    v8f acc = {};
#pragma unroll
    for (int ks = 0; ks < NKS; ++ks) {
      union { v16h v; v8h h[2]; } af;
      af.h[0] = *(const v8h*)(ap + ks * 32);
      af.h[1] = *(const v8h*)(ap + ks * 32 + 16);
      acc = __builtin_amdgcn_wmma_f32_16x16x32_f16(
          /*neg_a=*/false, af.v, /*neg_b=*/false, bf[ks],
          /*c_mod=*/(short)0, acc, /*reuse_a=*/false, /*reuse_b=*/false);
    }
#pragma unroll
    for (int r = 0; r < 8; ++r) {
      float val = acc[r] + bv;
      if (RELU) val = fmaxf(val, 0.f);
      op[r * NC] = (OUT_T)val;      // constant offset -> immediate in store
    }
    ap += aInc;
    op += oInc;
  }
}

// ---------------------------------------------------------------- driver

extern "C" void kernel_launch(void* const* d_in, const int* in_sizes, int n_in,
                              void* d_out, int out_size, void* d_ws, size_t ws_size,
                              hipStream_t stream) {
  const float* x        = (const float*)d_in[0];
  const int*   ei[3]    = {(const int*)d_in[1], (const int*)d_in[2], (const int*)d_in[3]};
  const float* in_gamma = (const float*)d_in[4];
  const float* in_beta  = (const float*)d_in[5];
  const float* W1       = (const float*)d_in[6];
  const float* b1       = (const float*)d_in[7];
  const float* W2       = (const float*)d_in[8];
  const float* b2       = (const float*)d_in[9];
  const float* bn_gamma = (const float*)d_in[10];
  const float* bn_beta  = (const float*)d_in[11];

  const int CIN = 128, COUT = 64;
  const int N = in_sizes[0] / CIN;   // 100000 (== 6250 * 16, tiles divide evenly)
  const int E = in_sizes[1] / 2;     // 1000000

  // workspace carve-out (~167 MB total)
  char* ws = (char*)d_ws;
  size_t off = 0;
  auto carve = [&](size_t bytes) -> void* {
    void* p = ws + off;
    off = (off + bytes + 255) & ~(size_t)255;
    return p;
  };
  float*    xn  = (float*)   carve((size_t)N * CIN * 4);
  float*    agg = (float*)   carve((size_t)N * CIN * 4);
  float*    h2  = (float*)   carve((size_t)N * COUT * 4);
  _Float16* h16 = (_Float16*)carve((size_t)N * CIN * 2);
  _Float16* t16 = (_Float16*)carve((size_t)N * COUT * 2);
  _Float16* w1s = (_Float16*)carve((size_t)CIN * COUT * 2);   // swizzled B, layer 1
  _Float16* w2s = (_Float16*)carve((size_t)COUT * COUT * 2);  // swizzled B, layer 2
  float*    st  = (float*)   carve(1024 * 4);
  float* inSum = st;        float* inSq    = st + 128;
  float* inScl = st + 256;  float* inShf   = st + 384;
  float* oSum  = st + 512;  float* oSq     = st + 576;  // contiguous 128 floats
  float* oScl  = st + 640;  float* oShf    = st + 704;

  const long totIn  = (long)N * CIN;
  const long totOut = (long)N * COUT;
  const float invN  = 1.f / (float)N;

  // ---- input BatchNorm (batch stats over node dim)
  zero_f32<<<1, 256, 0, stream>>>(inSum, 256);                       // inSum+inSq
  col_stats<<<1024, CIN, 0, stream>>>(x, N, CIN, inSum, inSq);
  bn_finalize<<<1, 128, 0, stream>>>(inSum, inSq, in_gamma, in_beta, inScl, inShf, CIN, invN);
  apply_affine128<<<2048, 256, 0, stream>>>(x, inScl, inShf, xn, totIn / 4);

  // GEMM launch geometry: 512 blocks * 8 waves = 4096 waves; a multiple of
  // NTILE=4 so every column tile gets the same wave-uniform row-tile stride.
  const int gemmBlocks = 512;

  for (int i = 0; i < 3; ++i) {
    // GINConv aggregation: agg = segment_sum(xn[src] -> dst)
    zero_f32_v4<<<2048, 256, 0, stream>>>(agg, totIn / 4);
    zero_f32<<<1, 128, 0, stream>>>(oSum, 128);                      // oSum+oSq
    scatter_add128<<<E, CIN, 0, stream>>>(xn, ei[i], ei[i] + E, agg, E);

    // h = xn + agg  (f16 for the WMMA path; post-GEMM BN re-normalizes)
    add_to_f16<<<2048, 256, 0, stream>>>(xn, agg, h16, totIn / 4);

    // per-branch weights -> f16, pre-swizzled into WMMA B-fragment order
    swizzle_B<<<32, 256, 0, stream>>>(W1 + (size_t)i * CIN * COUT, w1s, CIN, COUT);
    swizzle_B<<<16, 256, 0, stream>>>(W2 + (size_t)i * COUT * COUT, w2s, COUT, COUT);

    // MLP: relu(h @ W1 + b1) @ W2 + b2
    wmma_gemm<128, 64, true, _Float16><<<gemmBlocks, 256, 0, stream>>>(
        h16, w1s, b1 + i * COUT, t16, N);
    wmma_gemm<64, 64, false, float><<<gemmBlocks, 256, 0, stream>>>(
        t16, w2s, b2 + i * COUT, h2, N);

    // output BatchNorm + tanh -> d_out[i]
    col_stats<<<1024, COUT, 0, stream>>>(h2, N, COUT, oSum, oSq);
    bn_finalize<<<1, 64, 0, stream>>>(oSum, oSq, bn_gamma + i * COUT, bn_beta + i * COUT,
                                      oScl, oShf, COUT, invN);
    apply_affine_tanh64<<<1024, 256, 0, stream>>>(
        h2, oScl, oShf, (float*)d_out + (size_t)i * N * COUT, totOut / 4);
  }
}
